// mwt_transform_3487513444896
// MI455X (gfx1250) — matrix operand order repaired and checked by reading, not compiled, not run on hardware
//
#include <hip/hip_runtime.h>
#include <hip/hip_bf16.h>
#include <cstdint>

// ---------------------------------------------------------------------------
// MI455X (gfx1250) implementation of the multiwavelet spectral operator.
// Bandwidth-bound (402MB fp32 spectral weights reused over 10 levels):
//  - one-time fp32->bf16 transposed weight conversion into workspace
//  - all large GEMMs via v_wmma_f32_16x16x32_bf16 (f32 accumulate)
//  - per-mode complex GEMM K-split 4-ways (LDS reduction) so ~4 waves/SIMD
//    keep 23.3 TB/s of weight streaming in flight
//  - rFFT/irFFT with only l<=16 modes implemented as small DFTs (VALU)
// ---------------------------------------------------------------------------

typedef __attribute__((ext_vector_type(16))) __bf16 v16bf;
typedef __attribute__((ext_vector_type(8)))  float  v8f;

#define PI_F 3.14159265358979f

static constexpr int CK   = 1024;   // C*K channels
static constexpr int ICH_ = 512;
static constexpr int BATCH = 8;
static constexpr int LSEQ_ = 1024;

// ---------------- WMMA fragment builders (contiguous k order, same for A and B) ----------
// A (16x32, MxK): lane m = lane&15; element e -> K=k0+16*half+e
__device__ __forceinline__ v16bf make_a_frag_f32(const float* __restrict__ row, int k0, int half) {
  const float4* p0 = (const float4*)(row + k0 + 16 * half);
  const float4* p1 = (const float4*)(row + k0 + 16 * half + 8);
  float4 a0 = p0[0], a1 = p0[1], b0 = p1[0], b1 = p1[1];
  v16bf v;
  v[0]=(__bf16)a0.x; v[1]=(__bf16)a0.y; v[2]=(__bf16)a0.z; v[3]=(__bf16)a0.w;
  v[4]=(__bf16)a1.x; v[5]=(__bf16)a1.y; v[6]=(__bf16)a1.z; v[7]=(__bf16)a1.w;
  v[8]=(__bf16)b0.x; v[9]=(__bf16)b0.y; v[10]=(__bf16)b0.z; v[11]=(__bf16)b0.w;
  v[12]=(__bf16)b1.x; v[13]=(__bf16)b1.y; v[14]=(__bf16)b1.z; v[15]=(__bf16)b1.w;
  return v;
}
// B (32x16, KxN) with weights stored N-major ([n][k]): element e -> K=k0+16*half+e
__device__ __forceinline__ v16bf make_b_frag_f32(const float* __restrict__ nrow, int k0, int half) {
  const float4* p = (const float4*)(nrow + k0 + 16 * half);
  float4 q0 = p[0], q1 = p[1], q2 = p[2], q3 = p[3];
  v16bf v;
  v[0]=(__bf16)q0.x; v[1]=(__bf16)q0.y; v[2]=(__bf16)q0.z; v[3]=(__bf16)q0.w;
  v[4]=(__bf16)q1.x; v[5]=(__bf16)q1.y; v[6]=(__bf16)q1.z; v[7]=(__bf16)q1.w;
  v[8]=(__bf16)q2.x; v[9]=(__bf16)q2.y; v[10]=(__bf16)q2.z; v[11]=(__bf16)q2.w;
  v[12]=(__bf16)q3.x; v[13]=(__bf16)q3.y; v[14]=(__bf16)q3.z; v[15]=(__bf16)q3.w;
  return v;
}

// ---------------- GEMM: Out(MxN) = A(MxK) * W(NxK)^T + bias ---------------
// 8 waves / block, each wave owns a 16x16 tile.  Row-major fp32 in/out,
// on-the-fly bf16 conversion feeding v_wmma_f32_16x16x32_bf16.
__global__ void __launch_bounds__(256) gemm_wmma_nt(
    const float* __restrict__ A, const float* __restrict__ W,
    const float* __restrict__ bias, float* __restrict__ Out,
    int M, int N, int K) {
  const int lane = threadIdx.x & 31;
  const int wave = threadIdx.x >> 5;
  const int half = lane >> 4;
  const int nl   = lane & 15;
  const int n0 = blockIdx.x * 128 + wave * 16;
  const int m0 = blockIdx.y * 16;
  const float* arow = A + (size_t)(m0 + nl) * K;
  const float* wrow = W + (size_t)(n0 + nl) * K;
  v8f acc = {};
  for (int k0 = 0; k0 < K; k0 += 32) {
    v16bf af = make_a_frag_f32(arow, k0, half);
    v16bf bf = make_b_frag_f32(wrow, k0, half);
    acc = __builtin_amdgcn_wmma_f32_16x16x32_bf16(false, af, false, bf,
                                                  (short)0, acc, false, false);
  }
  const float bv = bias[n0 + nl];
  float* orow = Out + (size_t)(m0 + 8 * half) * N + n0 + nl;
  #pragma unroll
  for (int r = 0; r < 8; ++r) orow[(size_t)r * N] = acc[r] + bv;
}

// ---------------- spectral weight convert: w[i][o][x] -> bf16 [x][o][i] ----
__global__ void __launch_bounds__(256) spec_w_cvt(const float* __restrict__ W,
                                                  __bf16* __restrict__ Out) {
  const size_t idx = (size_t)blockIdx.x * 256 + threadIdx.x;  // 2^20 threads
  const int i = (int)(idx & 1023);
  const int o = (int)(idx >> 10);
  const float4* p = (const float4*)(W + ((size_t)i * 1024 + o) * 16);
  float v[16];
  float4 f;
  f = p[0]; v[0]=f.x;  v[1]=f.y;  v[2]=f.z;  v[3]=f.w;
  f = p[1]; v[4]=f.x;  v[5]=f.y;  v[6]=f.z;  v[7]=f.w;
  f = p[2]; v[8]=f.x;  v[9]=f.y;  v[10]=f.z; v[11]=f.w;
  f = p[3]; v[12]=f.x; v[13]=f.y; v[14]=f.z; v[15]=f.w;
  #pragma unroll
  for (int x = 0; x < 16; ++x)
    Out[((size_t)x << 20) + ((size_t)o << 10) + i] = (__bf16)v[x];
}

// ---------------- per-mode complex GEMM (the bandwidth hog) ----------------
// Inputs dfP/xfP packed [x][m][i] (m<8: Re of batch m, m>=8: Im of batch m-8).
// Computes Ud_f = df*A + xf*B and Us_f = df*C (complex), same packed layout.
// 4 waves/block, K split 4 ways (256 each), partial f32 tiles reduced in LDS.
__global__ void __launch_bounds__(128) mode_gemm_wmma(
    const float* __restrict__ dfP, const float* __restrict__ xfP,
    const __bf16* __restrict__ Ar, const __bf16* __restrict__ Ai,
    const __bf16* __restrict__ Br, const __bf16* __restrict__ Bi,
    const __bf16* __restrict__ Cr, const __bf16* __restrict__ Ci,
    float* __restrict__ UdP, float* __restrict__ UsP) {
  const int lane = threadIdx.x & 31;
  const int wave = threadIdx.x >> 5;
  const int half = lane >> 4;
  const int nl   = lane & 15;
  const int o0   = blockIdx.x * 16;
  const int x    = blockIdx.y;
  const size_t plane = (size_t)x << 20;        // x * 1024 * 1024
  const size_t pk    = (size_t)x * (16 * CK);  // packed-mode offset
  const float* drow = dfP + pk + (size_t)nl * CK;
  const float* xrow = xfP + pk + (size_t)nl * CK;
  const size_t orow = (size_t)(o0 + nl) * CK;
  const __bf16* pAr = Ar + plane + orow;
  const __bf16* pAi = Ai + plane + orow;
  const __bf16* pBr = Br + plane + orow;
  const __bf16* pBi = Bi + plane + orow;
  const __bf16* pCr = Cr + plane + orow;
  const __bf16* pCi = Ci + plane + orow;
  v8f aAr = {}, aAi = {}, aBr = {}, aBi = {}, aCr = {}, aCi = {};
  const int kbeg = wave * (CK / 4);
  const int kend = kbeg + (CK / 4);
  #pragma unroll 2
  for (int k0 = kbeg; k0 < kend; k0 += 32) {
    const int ko = k0 + 16 * half;
    __builtin_prefetch((const void*)(pAr + ko + 256), 0, 1);
    __builtin_prefetch((const void*)(pBr + ko + 256), 0, 1);
    __builtin_prefetch((const void*)(pCr + ko + 256), 0, 1);
    v16bf ad = make_a_frag_f32(drow, k0, half);
    v16bf ax = make_a_frag_f32(xrow, k0, half);
    v16bf bAr = *(const v16bf*)(pAr + ko);
    v16bf bAi = *(const v16bf*)(pAi + ko);
    v16bf bBr = *(const v16bf*)(pBr + ko);
    v16bf bBi = *(const v16bf*)(pBi + ko);
    v16bf bCr = *(const v16bf*)(pCr + ko);
    v16bf bCi = *(const v16bf*)(pCi + ko);
    aAr = __builtin_amdgcn_wmma_f32_16x16x32_bf16(false, ad, false, bAr, (short)0, aAr, false, false);
    aAi = __builtin_amdgcn_wmma_f32_16x16x32_bf16(false, ad, false, bAi, (short)0, aAi, false, false);
    aBr = __builtin_amdgcn_wmma_f32_16x16x32_bf16(false, ax, false, bBr, (short)0, aBr, false, false);
    aBi = __builtin_amdgcn_wmma_f32_16x16x32_bf16(false, ax, false, bBi, (short)0, aBi, false, false);
    aCr = __builtin_amdgcn_wmma_f32_16x16x32_bf16(false, ad, false, bCr, (short)0, aCr, false, false);
    aCi = __builtin_amdgcn_wmma_f32_16x16x32_bf16(false, ad, false, bCi, (short)0, aCi, false, false);
  }

  // ---- cross-wave K reduction through LDS: red[wave][acc][r][lane] ----
  __shared__ float red[4][6][8][32];
  #pragma unroll
  for (int r = 0; r < 8; ++r) {
    red[wave][0][r][lane] = aAr[r];
    red[wave][1][r][lane] = aAi[r];
    red[wave][2][r][lane] = aBr[r];
    red[wave][3][r][lane] = aBi[r];
    red[wave][4][r][lane] = aCr[r];
    red[wave][5][r][lane] = aCi[r];
  }
  __syncthreads();
  if (wave != 0) return;
  #pragma unroll
  for (int r = 0; r < 8; ++r) {
    aAr[r] = (red[0][0][r][lane] + red[1][0][r][lane]) + (red[2][0][r][lane] + red[3][0][r][lane]);
    aAi[r] = (red[0][1][r][lane] + red[1][1][r][lane]) + (red[2][1][r][lane] + red[3][1][r][lane]);
    aBr[r] = (red[0][2][r][lane] + red[1][2][r][lane]) + (red[2][2][r][lane] + red[3][2][r][lane]);
    aBi[r] = (red[0][3][r][lane] + red[1][3][r][lane]) + (red[2][3][r][lane] + red[3][3][r][lane]);
    aCr[r] = (red[0][4][r][lane] + red[1][4][r][lane]) + (red[2][4][r][lane] + red[3][4][r][lane]);
    aCi[r] = (red[0][5][r][lane] + red[1][5][r][lane]) + (red[2][5][r][lane] + red[3][5][r][lane]);
  }

  // complex recombine: packed row r+8*half gets (half==0 ? Re : Im)
  const float sgn = half ? 1.0f : -1.0f;
  float* ud = UdP + pk + (size_t)(8 * half) * CK + o0 + nl;
  float* us = UsP + pk + (size_t)(8 * half) * CK + o0 + nl;
  #pragma unroll
  for (int r = 0; r < 8; ++r) {
    float tA = __shfl_xor(aAi[r], 16, 32);
    float tB = __shfl_xor(aBi[r], 16, 32);
    float tC = __shfl_xor(aCi[r], 16, 32);
    ud[(size_t)r * CK] = (aAr[r] + sgn * tA) + (aBr[r] + sgn * tB);
    us[(size_t)r * CK] = aCr[r] + sgn * tC;
  }
}

// ---------------- forward DFT: y(b,t,i) -> packed [x][m][i] ---------------
__global__ void __launch_bounds__(256) dft_fwd(const float* __restrict__ Y,
                                               float* __restrict__ P, int n) {
  __shared__ float cs[512], sn[512];
  const int x = blockIdx.z, b = blockIdx.y;
  for (int t = threadIdx.x; t < n; t += 256) {
    float ang = -2.0f * PI_F * (float)(t * x) / (float)n;
    float s, c; __sincosf(ang, &s, &c);
    cs[t] = c; sn[t] = s;
  }
  __syncthreads();
  const int i = blockIdx.x * 256 + threadIdx.x;
  const float* yp = Y + (size_t)b * n * CK + i;
  float re = 0.f, im = 0.f;
  for (int t = 0; t < n; ++t) {
    float v = yp[(size_t)t * CK];
    re = fmaf(v, cs[t], re);
    im = fmaf(v, sn[t], im);
  }
  float* out = P + ((size_t)x * 16 + b) * CK + i;
  out[0] = re;
  out[8 * CK] = im;
}

// ---------------- inverse DFT (irfft, only l modes nonzero) ---------------
__global__ void __launch_bounds__(256) dft_inv(const float* __restrict__ P,
                                               float* __restrict__ Y, int n, int l) {
  const size_t idx = (size_t)blockIdx.x * 256 + threadIdx.x;
  const size_t total = (size_t)BATCH * n * CK;
  if (idx >= total) return;
  const int o = (int)(idx & 1023);
  const int t = (int)((idx >> 10) % (unsigned)n);
  const int b = (int)(idx / ((size_t)n << 10));
  const float w0 = 2.0f * PI_F * (float)t / (float)n;
  float acc = 0.f;
  for (int x = 0; x < l; ++x) {
    float wr = P[((size_t)x * 16 + b) * CK + o];
    float wi = P[((size_t)x * 16 + b + 8) * CK + o];
    float wgt = (x == 0 || 2 * x == n) ? 1.0f : 2.0f;
    float s, c; __sincosf(w0 * (float)x, &s, &c);
    acc = fmaf(wgt, wr * c - wi * s, acc);
  }
  Y[idx] = acc / (float)n;
}

// ---------------- wavelet split: xa=[x_even,x_odd]; d=xa*ec_d; xs=xa*ec_s --
__global__ void __launch_bounds__(256) wavelet_split(
    const float* __restrict__ X, const float* __restrict__ ecs,
    const float* __restrict__ ecd, float* __restrict__ Dd,
    float* __restrict__ Xs, int nh) {
  __shared__ float ss[128], sd[128];
  if (threadIdx.x < 128) { ss[threadIdx.x] = ecs[threadIdx.x]; sd[threadIdx.x] = ecd[threadIdx.x]; }
  __syncthreads();
  const size_t idx = (size_t)blockIdx.x * 256 + threadIdx.x;
  const size_t total = (size_t)BATCH * nh * 128;
  if (idx >= total) return;
  const int c = (int)(idx & 127);
  const int j = (int)((idx >> 7) % (unsigned)nh);
  const int b = (int)(idx / ((size_t)nh << 7));
  const float* pe = X + ((size_t)b * (2 * nh) + 2 * j) * CK + c * 8;
  float a[16];
  #pragma unroll
  for (int r = 0; r < 8; ++r) { a[r] = pe[r]; a[8 + r] = pe[CK + r]; }
  float* dp = Dd + ((size_t)b * nh + j) * CK + c * 8;
  float* xp = Xs + ((size_t)b * nh + j) * CK + c * 8;
  #pragma unroll
  for (int k = 0; k < 8; ++k) {
    float vd = 0.f, vs = 0.f;
    #pragma unroll
    for (int r = 0; r < 16; ++r) {
      vd = fmaf(a[r], sd[r * 8 + k], vd);
      vs = fmaf(a[r], ss[r * 8 + k], vs);
    }
    dp[k] = vd; xp[k] = vs;
  }
}

// ---------------- T0: x(b,1,c,k) @ T0_w^T + T0_b ---------------------------
__global__ void __launch_bounds__(256) t0_apply(const float* __restrict__ X,
                                                const float* __restrict__ Tw,
                                                const float* __restrict__ Tb,
                                                float* __restrict__ Y) {
  const int idx = blockIdx.x * 256 + threadIdx.x;  // 8*128 = 1024 (b,c) pairs
  if (idx >= BATCH * 128) return;
  const float* xp = X + (size_t)idx * 8;
  float a[8];
  #pragma unroll
  for (int r = 0; r < 8; ++r) a[r] = xp[r];
  float* yp = Y + (size_t)idx * 8;
  #pragma unroll
  for (int k = 0; k < 8; ++k) {
    float v = Tb[k];
    #pragma unroll
    for (int r = 0; r < 8; ++r) v = fmaf(a[r], Tw[k * 8 + r], v);
    yp[k] = v;
  }
}

// ---------------- reconstruction: x+=Us; cat(x,Ud); even/odd interleave ----
__global__ void __launch_bounds__(256) wavelet_recon(
    const float* __restrict__ X, const float* __restrict__ Us,
    const float* __restrict__ Ud, const float* __restrict__ rce,
    const float* __restrict__ rco, float* __restrict__ Y, int m) {
  __shared__ float se[128], so[128];
  if (threadIdx.x < 128) { se[threadIdx.x] = rce[threadIdx.x]; so[threadIdx.x] = rco[threadIdx.x]; }
  __syncthreads();
  const size_t idx = (size_t)blockIdx.x * 256 + threadIdx.x;
  const size_t total = (size_t)BATCH * m * 128;
  if (idx >= total) return;
  const int c = (int)(idx & 127);
  const int j = (int)((idx >> 7) % (unsigned)m);
  const int b = (int)(idx / ((size_t)m << 7));
  const size_t base = ((size_t)b * m + j) * CK + c * 8;
  float a[16];
  #pragma unroll
  for (int r = 0; r < 8; ++r) {
    a[r] = X[base + r] + Us[base + r];
    a[8 + r] = Ud[base + r];
  }
  float* ye = Y + ((size_t)b * (2 * m) + 2 * j) * CK + c * 8;
  #pragma unroll
  for (int k = 0; k < 8; ++k) {
    float ve = 0.f, vo = 0.f;
    #pragma unroll
    for (int r = 0; r < 16; ++r) {
      ve = fmaf(a[r], se[r * 8 + k], ve);
      vo = fmaf(a[r], so[r * 8 + k], vo);
    }
    ye[k] = ve;
    ye[CK + k] = vo;
  }
}

// ---------------------------------------------------------------------------
extern "C" void kernel_launch(void* const* d_in, const int* in_sizes, int n_in,
                              void* d_out, int out_size, void* d_ws, size_t ws_size,
                              hipStream_t stream) {
  (void)in_sizes; (void)n_in; (void)out_size; (void)ws_size;
  const float* values = (const float*)d_in[2];
  const float* Lk0_w  = (const float*)d_in[4];
  const float* Lk0_b  = (const float*)d_in[5];
  const float* Lk1_w  = (const float*)d_in[6];
  const float* Lk1_b  = (const float*)d_in[7];
  const float* T0_w   = (const float*)d_in[8];
  const float* T0_b   = (const float*)d_in[9];
  const float* ec_s   = (const float*)d_in[10];
  const float* ec_d   = (const float*)d_in[11];
  const float* rc_e   = (const float*)d_in[12];
  const float* rc_o   = (const float*)d_in[13];
  const float* spec[6] = {(const float*)d_in[14], (const float*)d_in[15],
                          (const float*)d_in[16], (const float*)d_in[17],
                          (const float*)d_in[18], (const float*)d_in[19]};

  // ---- workspace layout ----
  uint8_t* wsb = (uint8_t*)d_ws;
  size_t off = 0;
  auto alloc = [&](size_t bytes) -> void* {
    void* p = wsb + off;
    off += (bytes + 255) & ~(size_t)255;
    return p;
  };
  __bf16* wbf[6];
  for (int j = 0; j < 6; ++j) wbf[j] = (__bf16*)alloc((size_t)16 * 1024 * 1024 * sizeof(__bf16));
  float* X0  = (float*)alloc((size_t)BATCH * LSEQ_ * CK * 4);
  float* X1  = (float*)alloc((size_t)BATCH * LSEQ_ * CK * 4);
  float* Db  = (float*)alloc((size_t)BATCH * 512 * CK * 4);
  float* UD  = (float*)alloc((size_t)BATCH * 1023 * CK * 4);
  float* US  = (float*)alloc((size_t)BATCH * 1023 * CK * 4);
  float* dfP = (float*)alloc((size_t)16 * 16 * CK * 4);
  float* xfP = (float*)alloc((size_t)16 * 16 * CK * 4);
  float* UdP = (float*)alloc((size_t)16 * 16 * CK * 4);
  float* UsP = (float*)alloc((size_t)16 * 16 * CK * 4);

  // ---- 0. convert spectral weights (w[i][o][x] fp32 -> [x][o][i] bf16) ----
  for (int j = 0; j < 6; ++j)
    spec_w_cvt<<<4096, 256, 0, stream>>>(spec[j], wbf[j]);

  // ---- 1. V0 = values @ Lk0_w^T + Lk0_b  (M=8192,N=1024,K=512) ----
  gemm_wmma_nt<<<dim3(CK / 128, (BATCH * LSEQ_) / 16), 256, 0, stream>>>(
      values, Lk0_w, Lk0_b, X0, BATCH * LSEQ_, CK, ICH_);

  float* xcur[2] = {X0, X1};

  // ---- 2. decomposition levels ----
  for (int lvl = 0; lvl < 10; ++lvl) {
    const int n  = LSEQ_ >> lvl;
    const int nh = n >> 1;
    const int l  = (nh / 2 + 1 < 16) ? (nh / 2 + 1) : 16;
    float* xi = xcur[lvl & 1];
    float* xo = xcur[(lvl + 1) & 1];
    const size_t uoff = (size_t)BATCH * CK * (size_t)(1024 - (1024 >> lvl));

    const int tsplit = BATCH * nh * 128;
    wavelet_split<<<(tsplit + 255) / 256, 256, 0, stream>>>(xi, ec_s, ec_d, Db, xo, nh);

    dim3 gf(CK / 256, BATCH, l);
    dft_fwd<<<gf, 256, 0, stream>>>(Db, dfP, nh);
    dft_fwd<<<gf, 256, 0, stream>>>(xo, xfP, nh);

    mode_gemm_wmma<<<dim3(CK / 16, l), 128, 0, stream>>>(
        dfP, xfP, wbf[0], wbf[1], wbf[2], wbf[3], wbf[4], wbf[5], UdP, UsP);

    const int tinv = BATCH * nh * CK;
    dft_inv<<<(tinv + 255) / 256, 256, 0, stream>>>(UdP, UD + uoff, nh, l);
    dft_inv<<<(tinv + 255) / 256, 256, 0, stream>>>(UsP, US + uoff, nh, l);
  }

  // ---- 3. T0 on the length-1 remainder (sits in X0 after 10 levels) ----
  t0_apply<<<4, 256, 0, stream>>>(X0, T0_w, T0_b, X1);

  // ---- 4. reconstruction ----
  float* cur = X1;
  float* nxt = X0;
  for (int lvl = 9; lvl >= 0; --lvl) {
    const int m = LSEQ_ >> (lvl + 1);
    const size_t uoff = (size_t)BATCH * CK * (size_t)(1024 - (1024 >> lvl));
    const int trec = BATCH * m * 128;
    wavelet_recon<<<(trec + 255) / 256, 256, 0, stream>>>(
        cur, US + uoff, UD + uoff, rc_e, rc_o, nxt, m);
    float* t = cur; cur = nxt; nxt = t;
  }
  // final x (8,1024,1024) is in `cur`

  // ---- 5. out = x @ Lk1_w^T + Lk1_b  (M=8192,N=512,K=1024) ----
  gemm_wmma_nt<<<dim3(ICH_ / 128, (BATCH * LSEQ_) / 16), 256, 0, stream>>>(
      cur, Lk1_w, Lk1_b, (float*)d_out, BATCH * LSEQ_, ICH_, CK);
}
